// MultiHeadAttention_4157528343277
// MI455X (gfx1250) — matrix operand order repaired and checked by reading, not compiled, not run on hardware
//
#include <hip/hip_runtime.h>

typedef _Float16 f16;
typedef __attribute__((ext_vector_type(16))) _Float16 v16h;
typedef __attribute__((ext_vector_type(8)))  float    v8f;
typedef __attribute__((ext_vector_type(4)))  unsigned int u32x4;

#define NB 4
#define NS 2048
#define ND 1024
#define NH 16
#define NHD 64

// ---- WMMA operand loaders (contiguous k order, same for A and B) -------------------
// A (16x32 f16): lane holds row M=lane&15; contiguous K-run of 16 at (lane>=16?16:0).
__device__ __forceinline__ v16h load_a16(const f16* p0, const f16* p1) {
  union { v16h v; u32x4 q[2]; } u;
  u.q[0] = *(const u32x4*)p0;
  u.q[1] = *(const u32x4*)(p1);
  return u.v;
}
// B (32x16 f16): lane holds column N=lane&15; contiguous K-run of 16 starting
// at (lane>=16 ? 16 : 0).
__device__ __forceinline__ v16h load_b16(const f16* p) {
  union { v16h v; u32x4 q[2]; } u;
  u.q[0] = *(const u32x4*)p;
  u.q[1] = *(const u32x4*)(p + 8);
  return u.v;
}
__device__ __forceinline__ v8f wmma_f16(v16h a, v16h b, v8f c) {
  return __builtin_amdgcn_wmma_f32_16x16x32_f16(false, a, false, b,
                                                (short)0, c, false, false);
}
// LDS byte address = low 32 bits of flat address (ISA 10.2 aperture mapping)
__device__ __forceinline__ unsigned int lds_addr(const void* p) {
  return (unsigned int)(unsigned long long)p;
}
// gfx1250 async global->LDS copy, tracked by ASYNCcnt (ISA 10.x / 15.18.3)
#define ASYNC_CP_B128(ldsdst, gsrc)                                        \
  asm volatile("global_load_async_to_lds_b128 %0, %1, off"                 \
               :: "v"(ldsdst), "v"(gsrc) : "memory")
#define WAIT_ASYNC(n)                                                      \
  asm volatile("s_wait_asynccnt %0" :: "i"(n) : "memory")

// ---- elementwise f32 -> f16 ------------------------------------------------
__global__ __launch_bounds__(256) void cvt_f32_f16(const float* __restrict__ in,
                                                   f16* __restrict__ out, int n) {
  int i = blockIdx.x * 256 + threadIdx.x;
  if (i < n) out[i] = (f16)in[i];
}

// ---- tiled transpose + convert: W[K=1024][N=1024] f32 -> WT[N][K] f16 ------
__global__ __launch_bounds__(256) void xpose_w(const float* __restrict__ W,
                                               f16* __restrict__ WT) {
  __shared__ f16 tile[32][33];
  const int tx = threadIdx.x & 31, ty = threadIdx.x >> 5;
  const int bn = blockIdx.x * 32;  // N base
  const int bk = blockIdx.y * 32;  // K base
  #pragma unroll
  for (int i = 0; i < 32; i += 8)
    tile[ty + i][tx] = (f16)W[(size_t)(bk + ty + i) * ND + (bn + tx)];
  __syncthreads();
  #pragma unroll
  for (int i = 0; i < 32; i += 8)
    WT[(size_t)(bn + ty + i) * ND + (bk + tx)] = tile[tx][ty + i];
}

// ---- WMMA GEMM: C[M=8192, N=1024] = A_f16[M,K=1024] * BT_f16[N,K]^T + bias -
// Each wave computes a 16x64 output strip (4 accum tiles), K-step 32.
// mode 0: out f16 head-split  [B,H,S,HD]   (Q, K projections)
// mode 1: out f16 transposed  [B,H,HD,S]   (V projection)
// mode 2: out f32 row-major   [M,N]        (final output projection)
__global__ __launch_bounds__(256) void gemm_wmma(const f16* __restrict__ A,
                                                 const f16* __restrict__ BT,
                                                 const float* __restrict__ bias,
                                                 void* __restrict__ out, int mode) {
  const int K = ND;
  const int lane = threadIdx.x & 31;
  const int wave = (blockIdx.x * 256 + threadIdx.x) >> 5;
  const int nTiles = ND / 64;            // 16
  const int mi = wave / nTiles;          // 0..511
  const int nj = wave % nTiles;          // 0..15
  const int m0 = mi * 16, n0 = nj * 64;
  const int row = lane & 15, hi = lane >> 4;

  const f16* arow = A + (size_t)(m0 + row) * K;
  v8f acc[4];
  #pragma unroll
  for (int t = 0; t < 4; ++t)
    #pragma unroll
    for (int r = 0; r < 8; ++r) acc[t][r] = 0.0f;

  for (int k = 0; k < K; k += 32) {
    const int c0 = k + hi * 8;
    v16h a = load_a16(arow + k + hi * 16, arow + k + hi * 16 + 8);
    #pragma unroll
    for (int t = 0; t < 4; ++t) {
      const f16* brow = BT + (size_t)(n0 + t * 16 + row) * K;
      v16h b = load_b16(brow + k + hi * 16);
      acc[t] = wmma_f16(a, b, acc[t]);
    }
  }
  #pragma unroll
  for (int t = 0; t < 4; ++t) {
    float bv = bias[n0 + t * 16 + row];
    #pragma unroll
    for (int r = 0; r < 8; ++r) acc[t][r] += bv;
  }
  if (mode == 2) {
    float* o = (float*)out;
    #pragma unroll
    for (int t = 0; t < 4; ++t)
      #pragma unroll
      for (int r = 0; r < 8; ++r)
        o[(size_t)(m0 + r + hi * 8) * ND + (n0 + t * 16 + row)] = acc[t][r];
  } else {
    f16* o = (f16*)out;
    #pragma unroll
    for (int t = 0; t < 4; ++t) {
      const int n = n0 + t * 16 + row;
      const int hh = n >> 6, hd = n & 63;
      #pragma unroll
      for (int r = 0; r < 8; ++r) {
        const int m = m0 + r + hi * 8;
        const int bb = m >> 11, s = m & (NS - 1);
        size_t idx = (mode == 0)
            ? (((size_t)(bb * NH + hh) * NS + s) * NHD + hd)
            : (((size_t)(bb * NH + hh) * NHD + hd) * NS + s);
        o[idx] = (f16)acc[t][r];
      }
    }
  }
}

// ---- fused flash attention -------------------------------------------------
// Block = 8 waves on ONE (b,h), 128-query strip. K/V tiles (32 keys) staged
// into double-buffered LDS with gfx1250 async-to-LDS copies shared by all
// 8 waves (8x less K/V read traffic than per-wave streaming).
__global__ __launch_bounds__(256) void mha_flash(const f16* __restrict__ Qh,
                                                 const f16* __restrict__ Kh,
                                                 const f16* __restrict__ Vt,
                                                 const float* __restrict__ mask,
                                                 f16* __restrict__ ctx) {
  __shared__ __align__(16) f16 Ksh[2][32][72];  // [buf][key][hd]  (+8 pad)
  __shared__ __align__(16) f16 Vsh[2][64][40];  // [buf][hd][key]  (+8 pad)
  __shared__ __align__(16) f16 P[8][16][40];    // per-wave prob tile (+pad)

  const int tid = threadIdx.x;
  const int lane = tid & 31;
  const int wv = tid >> 5;
  const int bh = blockIdx.x >> 4;               // 0..63
  const int strip = blockIdx.x & 15;            // 16 strips of 128 queries
  const int b = bh >> 4, h = bh & 15;
  const int q0 = strip * 128 + wv * 16;
  const int row = lane & 15, hi = lane >> 4;
  const float LOG2E = 1.4426950408889634f;

  const f16* Qb = Qh + ((size_t)bh * NS + q0) * NHD;
  const f16* Kb = Kh + (size_t)bh * NS * NHD;
  const f16* Vb = Vt + (size_t)bh * NHD * NS;
  const float* mb = mask + (size_t)b * NS;

  // per-thread async-copy slice: K tile = 32 rows x 8 x b128, V = 64 x 4 x b128
  const int kk = tid >> 3, kpart = tid & 7;
  const int vh = tid >> 2, vpart = tid & 3;
  const f16* ksrc = Kb + (size_t)kk * NHD + kpart * 8;   // + j*32*NHD per block
  const f16* vsrc = Vb + (size_t)vh * NS + vpart * 8;    // + j*32     per block
  const unsigned int kdst0 = lds_addr(&Ksh[0][kk][kpart * 8]);
  const unsigned int kdst1 = lds_addr(&Ksh[1][kk][kpart * 8]);
  const unsigned int vdst0 = lds_addr(&Vsh[0][vh][vpart * 8]);
  const unsigned int vdst1 = lds_addr(&Vsh[1][vh][vpart * 8]);

  // Q A-operands (loaded once per wave)
  const f16* qrow = Qb + row * NHD;
  const v16h aQ0 = load_a16(qrow + hi * 16,      qrow + hi * 16 + 8);
  const v16h aQ1 = load_a16(qrow + 32 + hi * 16, qrow + 32 + hi * 16 + 8);

  float m[8], l[8];
  v8f acc[4];
  #pragma unroll
  for (int r = 0; r < 8; ++r) { m[r] = -3.0e38f; l[r] = 0.0f; }
  #pragma unroll
  for (int t = 0; t < 4; ++t)
    #pragma unroll
    for (int r = 0; r < 8; ++r) acc[t][r] = 0.0f;

  // prologue: stage key-block 0 into buffer 0
  ASYNC_CP_B128(kdst0, ksrc);
  ASYNC_CP_B128(vdst0, vsrc);

  const int NKB = NS / 32;                      // 64 key blocks
  for (int j = 0; j < NKB; ++j) {
    // issue copies for block j+1 into the other buffer, then wait for block j
    if (j + 1 < NKB) {
      const f16* kn = ksrc + (size_t)(j + 1) * 32 * NHD;
      const f16* vn = vsrc + (j + 1) * 32;
      if ((j + 1) & 1) { ASYNC_CP_B128(kdst1, kn); ASYNC_CP_B128(vdst1, vn); }
      else             { ASYNC_CP_B128(kdst0, kn); ASYNC_CP_B128(vdst0, vn); }
      WAIT_ASYNC(2);
    } else {
      WAIT_ASYNC(0);
    }
    __syncthreads();                            // all slices of block j visible
    const int buf = j & 1;
    const int kb = j * 32;

    // ---- scores: S = Q * K^T over 32 keys (two 16x16 tiles), K from LDS ----
    v8f s0, s1;
    #pragma unroll
    for (int r = 0; r < 8; ++r) { s0[r] = 0.0f; s1[r] = 0.0f; }
    s0 = wmma_f16(aQ0, load_b16(&Ksh[buf][row][hi * 16]),           s0);
    s0 = wmma_f16(aQ1, load_b16(&Ksh[buf][row][32 + hi * 16]),      s0);
    s1 = wmma_f16(aQ0, load_b16(&Ksh[buf][16 + row][hi * 16]),      s1);
    s1 = wmma_f16(aQ1, load_b16(&Ksh[buf][16 + row][32 + hi * 16]), s1);

    const float mk0 = (1.0f - mb[kb + row]) * -1e9f;
    const float mk1 = (1.0f - mb[kb + 16 + row]) * -1e9f;
    #pragma unroll
    for (int r = 0; r < 8; ++r) {
      s0[r] = s0[r] * 0.125f + mk0;             // 1/sqrt(64)
      s1[r] = s1[r] * 0.125f + mk1;
    }
    // ---- online softmax (row = one VGPR across a 16-lane half) ----
    #pragma unroll
    for (int r = 0; r < 8; ++r) {
      float v = fmaxf(s0[r], s1[r]);
      #pragma unroll
      for (int off = 8; off >= 1; off >>= 1) v = fmaxf(v, __shfl_xor(v, off, 32));
      const float mn = fmaxf(m[r], v);
      const float al = __builtin_amdgcn_exp2f((m[r] - mn) * LOG2E);
      m[r] = mn;
      const float p0 = __builtin_amdgcn_exp2f((s0[r] - mn) * LOG2E);
      const float p1 = __builtin_amdgcn_exp2f((s1[r] - mn) * LOG2E);
      s0[r] = p0; s1[r] = p1;
      float rs = p0 + p1;
      #pragma unroll
      for (int off = 8; off >= 1; off >>= 1) rs += __shfl_xor(rs, off, 32);
      l[r] = l[r] * al + rs;
      acc[0][r] *= al; acc[1][r] *= al; acc[2][r] *= al; acc[3][r] *= al;
    }
    // ---- C-layout -> A-layout via wave-private LDS tile ----
    #pragma unroll
    for (int r = 0; r < 8; ++r) {
      P[wv][r + hi * 8][row]      = (f16)s0[r];
      P[wv][r + hi * 8][16 + row] = (f16)s1[r];
    }
    const f16* prow = &P[wv][row][0];
    const v16h aP = load_a16(prow + hi * 16, prow + hi * 16 + 8);
    // ---- O += P * V, V tile from LDS (transposed [hd][key]) ----
    #pragma unroll
    for (int t = 0; t < 4; ++t)
      acc[t] = wmma_f16(aP, load_b16(&Vsh[buf][t * 16 + row][hi * 16]), acc[t]);

    __syncthreads();   // all waves done reading buf before it is refilled (j+2)
  }
  // ---- normalize + store context f16 [B,S,D] with head interleave ----
  #pragma unroll
  for (int r = 0; r < 8; ++r) {
    const float inv = 1.0f / l[r];
    acc[0][r] *= inv; acc[1][r] *= inv; acc[2][r] *= inv; acc[3][r] *= inv;
  }
  #pragma unroll
  for (int t = 0; t < 4; ++t)
    #pragma unroll
    for (int r = 0; r < 8; ++r)
      ctx[((size_t)b * NS + q0 + r + hi * 8) * ND + h * 64 + t * 16 + row] =
          (f16)acc[t][r];
}

// ---- host launch -----------------------------------------------------------
extern "C" void kernel_launch(void* const* d_in, const int* in_sizes, int n_in,
                              void* d_out, int out_size, void* d_ws, size_t ws_size,
                              hipStream_t stream) {
  const float* query = (const float*)d_in[0];
  const float* key   = (const float*)d_in[1];
  const float* value = (const float*)d_in[2];
  const float* mask  = (const float*)d_in[3];
  const float* wq = (const float*)d_in[4];  const float* bq = (const float*)d_in[5];
  const float* wk = (const float*)d_in[6];  const float* bk = (const float*)d_in[7];
  const float* wv = (const float*)d_in[8];  const float* bv = (const float*)d_in[9];
  const float* wo = (const float*)d_in[10]; const float* bo = (const float*)d_in[11];

  char* ws = (char*)d_ws;
  const size_t XSZ = (size_t)NB * NS * ND * sizeof(f16);  // 16 MB
  const size_t WSZ = (size_t)ND * ND * sizeof(f16);       // 2 MB
  f16* Xq  = (f16*)(ws);
  f16* Xk  = (f16*)(ws + XSZ);
  f16* Xv  = (f16*)(ws + 2 * XSZ);
  f16* WTq = (f16*)(ws + 3 * XSZ);
  f16* WTk = (f16*)(ws + 3 * XSZ + WSZ);
  f16* WTv = (f16*)(ws + 3 * XSZ + 2 * WSZ);
  f16* WTo = (f16*)(ws + 3 * XSZ + 3 * WSZ);
  f16* Qh  = (f16*)(ws + 3 * XSZ + 4 * WSZ);
  f16* Kh  = (f16*)(ws + 4 * XSZ + 4 * WSZ);
  f16* Vt  = (f16*)(ws + 5 * XSZ + 4 * WSZ);
  f16* Ctx = (f16*)(ws + 6 * XSZ + 4 * WSZ);   // total ~120 MB

  const int NEL = NB * NS * ND;                // 8388608
  cvt_f32_f16<<<NEL / 256, 256, 0, stream>>>(query, Xq, NEL);
  cvt_f32_f16<<<NEL / 256, 256, 0, stream>>>(key,   Xk, NEL);
  cvt_f32_f16<<<NEL / 256, 256, 0, stream>>>(value, Xv, NEL);

  dim3 tg(ND / 32, ND / 32);
  xpose_w<<<tg, 256, 0, stream>>>(wq, WTq);
  xpose_w<<<tg, 256, 0, stream>>>(wk, WTk);
  xpose_w<<<tg, 256, 0, stream>>>(wv, WTv);
  xpose_w<<<tg, 256, 0, stream>>>(wo, WTo);

  gemm_wmma<<<1024, 256, 0, stream>>>(Xq, WTq, bq, (void*)Qh, 0);
  gemm_wmma<<<1024, 256, 0, stream>>>(Xk, WTk, bk, (void*)Kh, 0);
  gemm_wmma<<<1024, 256, 0, stream>>>(Xv, WTv, bv, (void*)Vt, 1);

  // 64 (b,h) x 16 query strips = 1024 blocks, 8 waves each
  mha_flash<<<1024, 256, 0, stream>>>(Qh, Kh, Vt, mask, Ctx);

  gemm_wmma<<<1024, 256, 0, stream>>>(Ctx, WTo, bo, d_out, 2);
}
